// PGLSTM_20607253086781
// MI455X (gfx1250) — matrix operand order repaired and checked by reading, not compiled, not run on hardware
//
#include <hip/hip_runtime.h>

// ---------------------------------------------------------------------------
// PG-LSTM (hypernetwork-generated per-example LSTM) for MI455X / gfx1250.
//
// Shapes: B=64, T=128, H=IN=128, L=2, CTX=CH=512, WROWS=256, GATES=512.
//
// Pipeline (all on `stream`, serialized by stream order):
//   K0: context f32 -> f16                                   (tiny)
//   K1: Hw16[l] = f16(relu(ctx @ W1w[l]))   [WMMA]           (tiny)
//       Hb16[l] = f16(relu(ctx @ W1b[l]))   [WMMA]
//       Bias[l] = Hb16[l] @ W2b[l]          [WMMA, f32 out]
//   K2: Wgen = Hw16[l] @ W2w[l]             [WMMA + TDM stage]  HBM-bound
//       scatter rows 0..127 -> Wx16 (input weights, f16)
//               rows 128..255 -> Wh16 (recurrent weights, f16, k-pair packed)
//   K3: XG[b] = X[b] @ Wx16[b] + bias       [WMMA]  (time-batched input proj)
//   K4: persistent scan: Wh16[b] -> LDS via TDM (128 KB), 128 sequential
//       steps of 512-wide GEMV + gate nonlinearities.        (VALU/LDS)
// ---------------------------------------------------------------------------

typedef __attribute__((ext_vector_type(16))) _Float16     v16h;
typedef __attribute__((ext_vector_type(8)))  float        v8f;
typedef __attribute__((ext_vector_type(4)))  unsigned int v4u;
typedef __attribute__((ext_vector_type(8)))  int          v8i;
typedef __attribute__((ext_vector_type(4)))  int          v4i;

#define BB   64
#define TT   128
#define HH   128
#define CTX  512
#define CH   512
#define GATES 512
#define NW2  131072           // WROWS * GATES

__device__ __forceinline__ v8f wmma16(v16h a, v16h b, v8f c) {
  return __builtin_amdgcn_wmma_f32_16x16x32_f16(
      /*neg_a=*/false, a, /*neg_b=*/false, b,
      /*c_mod=*/(short)0, c, /*reuse_a=*/false, /*reuse_b=*/false);
}

// ---------------------------------------------------------------------------
// Tensor Data Mover: contiguous 1-D global -> LDS copy expressed as a
// degenerate 2-D tile.  D# per CDNA5 ISA:
//   group0: count=1 | lds_addr | global_addr[31:0] | global_addr[56:32]+type=2
//   group1: data_size=3 (8B), tile_dim0 = tensor_dim0 = stride0 = bytes/8,
//           tile_dim1 = tensor_dim1 = 1, workgroup_mask = 0 (no cluster).
// `bytes` must be a multiple of 8 and < 512 KB (tile_dim0 is 16 bits).
// Issue from ONE wave only; TDM ignores EXEC, uses TENSORcnt.
// This toolchain exposes the 6-arg builtin form:
//   (uint32x4 g0, int32x8 g1, int32x4 g2, int32x4 g3, int32x8 extra, i32 cpol)
__device__ __forceinline__ void tdm_load_1d(void* lds_dst, const void* gsrc,
                                            unsigned bytes) {
  const unsigned elems = bytes >> 3;                       // 8-byte units
  const unsigned lds   = (unsigned)(size_t)lds_dst;        // addr[31:0] = LDS offset
  const unsigned long long ga = (unsigned long long)(size_t)gsrc;
  v4u g0 = { 1u,                                           // count=1, user mode
             lds,
             (unsigned)(ga & 0xffffffffu),
             (unsigned)((ga >> 32) & 0x01ffffffu) | (2u << 30) };   // type=2
  v8i g1 = { (int)0x00030000u,                             // data_size=3, mask=0
             (int)((elems & 0xffffu) << 16),               // tensor_dim0[15:0]
             (int)(((elems >> 16) & 0xffffu) | (1u << 16)),// dim0[31:16]|tensor_dim1=1
             (int)((elems & 0xffffu) << 16),               // tile_dim0
             (int)1,                                       // tile_dim1=1
             (int)elems,                                   // tensor_dim0_stride[31:0]
             (int)((elems >> 16) & 0xffffu),               // stride[47:32]
             0 };
  v4i z4 = { 0, 0, 0, 0 };
  v8i z8 = { 0, 0, 0, 0, 0, 0, 0, 0 };
  __builtin_amdgcn_tensor_load_to_lds(g0, g1, z4, z4, z8, 0);
}

// A fragment: 16x32 f16 tile at (m0,k0) of row-major A[ldA].
// This code's contiguous k order (same as B): lanes 0-15 hold K=k0..k0+15; lanes 16-31 hold K=k0+16..k0+31.
__device__ __forceinline__ v16h load_a_frag(const _Float16* A, int ldA,
                                            int m0, int k0, int lane) {
  const int hl  = lane >> 4;
  const int row = m0 + (lane & 15);
  const int kb  = k0 + hl * 16;
  v16h a;
#pragma unroll
  for (int e = 0; e < 8; ++e) a[e]     = A[row * ldA + kb + e];
#pragma unroll
  for (int e = 0; e < 8; ++e) a[8 + e] = A[row * ldA + kb + 8 + e];
  return a;
}

// B fragment: 32x16 tile at (k0,n0) of row-major B[ldB], f32 source (convert).
// This code's contiguous k order (same as A): lane = column; lanes 0-15 hold K=k0..k0+15, lanes 16-31 K+16.
__device__ __forceinline__ v16h load_b_frag_f32(const float* B, int ldB,
                                                int k0, int n0, int lane) {
  const int hl  = lane >> 4;
  const int col = n0 + (lane & 15);
  const int kb  = k0 + hl * 16;
  v16h b;
#pragma unroll
  for (int e = 0; e < 16; ++e) b[e] = (_Float16)B[(kb + e) * ldB + col];
  return b;
}

__device__ __forceinline__ v16h load_b_frag_f16(const _Float16* B, int ldB,
                                                int k0, int n0, int lane) {
  const int hl  = lane >> 4;
  const int col = n0 + (lane & 15);
  const int kb  = k0 + hl * 16;
  v16h b;
#pragma unroll
  for (int e = 0; e < 16; ++e) b[e] = B[(kb + e) * ldB + col];
  return b;
}

// ---------------------------------------------------------------------------
// K0: f32 -> f16 convert
__global__ void k_cvt_f16(const float* __restrict__ src,
                          _Float16* __restrict__ dst, int n) {
  int i = blockIdx.x * blockDim.x + threadIdx.x;
  if (i < n) dst[i] = (_Float16)src[i];
}

// ---------------------------------------------------------------------------
// K1: D[64,512] = act(A16[64,K] @ B32[K,512]).  grid=8, block=128 (4 waves).
template <bool RELU, bool OUT16>
__global__ void k_small_gemm(const _Float16* __restrict__ A,
                             const float* __restrict__ B,
                             void* __restrict__ Dout, int K) {
  const int lane = threadIdx.x & 31;
  const int wave = threadIdx.x >> 5;
  const int n0   = (blockIdx.x * 4 + wave) * 16;

  v8f c0 = {}, c1 = {}, c2 = {}, c3 = {};
  for (int k0 = 0; k0 < K; k0 += 32) {
    v16h bf = load_b_frag_f32(B, 512, k0, n0, lane);
    c0 = wmma16(load_a_frag(A, K,  0, k0, lane), bf, c0);
    c1 = wmma16(load_a_frag(A, K, 16, k0, lane), bf, c1);
    c2 = wmma16(load_a_frag(A, K, 32, k0, lane), bf, c2);
    c3 = wmma16(load_a_frag(A, K, 48, k0, lane), bf, c3);
  }
  const int col = n0 + (lane & 15);
  const int hl  = lane >> 4;
  v8f acc[4] = {c0, c1, c2, c3};
#pragma unroll
  for (int mt = 0; mt < 4; ++mt) {
#pragma unroll
    for (int v = 0; v < 8; ++v) {
      float val = acc[mt][v];
      if (RELU) val = fmaxf(val, 0.0f);
      int m = mt * 16 + hl * 8 + v;
      if (OUT16) ((_Float16*)Dout)[m * 512 + col] = (_Float16)val;
      else       ((float*)Dout)[m * 512 + col] = val;
    }
  }
}

// ---------------------------------------------------------------------------
// K2: Wgen[l] = Hw16[l] @ W2w[l]  ([64,512] x [512,131072], HBM-bound).
// grid=(1024, L), block=256 (8 waves, one 16-col N-tile each).
// A staged to LDS by the Tensor Data Mover (64 KB contiguous).
__global__ void k_gen_w(const _Float16* __restrict__ Hw16,
                        const float* __restrict__ W2w,
                        _Float16* __restrict__ Wx16,
                        _Float16* __restrict__ Wh16) {
  __shared__ _Float16 As[64 * 512];           // 64 KB activation stage
  const int l    = blockIdx.y;
  const int tid  = threadIdx.x;
  const int lane = tid & 31;
  const int wave = tid >> 5;

  if (wave == 0) {                            // one wave issues the DMA
    tdm_load_1d(As, Hw16 + (size_t)l * 64 * 512, 64 * 512 * 2);
    __builtin_amdgcn_s_wait_tensorcnt(0);
  }
  __syncthreads();

  const float* Bg = W2w + (size_t)l * 512 * NW2;
  const int n0 = (blockIdx.x * 8 + wave) * 16;

  v8f c0 = {}, c1 = {}, c2 = {}, c3 = {};
  for (int k0 = 0; k0 < 512; k0 += 32) {
    if (k0 + 32 < 512)   // keep the HBM stream ahead: global_prefetch_b8
      __builtin_prefetch(Bg + (size_t)(k0 + 32) * NW2 + n0 + lane, 0, 1);
    v16h bf = load_b_frag_f32(Bg, NW2, k0, n0, lane);
    c0 = wmma16(load_a_frag(As, 512,  0, k0, lane), bf, c0);
    c1 = wmma16(load_a_frag(As, 512, 16, k0, lane), bf, c1);
    c2 = wmma16(load_a_frag(As, 512, 32, k0, lane), bf, c2);
    c3 = wmma16(load_a_frag(As, 512, 48, k0, lane), bf, c3);
  }

  const int r  = n0 >> 9;                       // weight row 0..255
  const int j  = (n0 & 511) + (lane & 15);      // gate column 0..511
  const int hl = lane >> 4;
  v8f acc[4] = {c0, c1, c2, c3};
#pragma unroll
  for (int mt = 0; mt < 4; ++mt) {
#pragma unroll
    for (int v = 0; v < 8; ++v) {
      const int b = mt * 16 + hl * 8 + v;       // batch index (M dim)
      const _Float16 val = (_Float16)acc[mt][v];
      if (r < 128) {
        Wx16[(((size_t)l * 64 + b) * 128 + r) * 512 + j] = val;
      } else {
        const int rp = r - 128;
        Wh16[(((((size_t)l * 64 + b) * 64 + (rp >> 1)) * 512 + j) << 1) + (rp & 1)] = val;
      }
    }
  }
}

// ---------------------------------------------------------------------------
// K3: XG[b] = Xin[b] @ Wx16[b] + bias   ([128,128] x [128,512] per example).
// grid=(2, 64), block=512 (16 waves, one N-tile each).
__global__ void k_inproj(const float* __restrict__ Xin,          // [64][128][128]
                         const _Float16* __restrict__ Wx16_l,    // [64][128][512]
                         const float* __restrict__ Bias_l,       // [64][512]
                         float* __restrict__ XG) {               // [64][128][512]
  __shared__ _Float16 Axs[128 * 128];          // 32 KB, f16 copy of X[b]
  const int b   = blockIdx.y;
  const int tid = threadIdx.x;

  const float* xg = Xin + (size_t)b * 128 * 128;
  for (int i = tid; i < 128 * 128; i += 512) Axs[i] = (_Float16)xg[i];
  __syncthreads();

  const int lane = tid & 31;
  const int wave = tid >> 5;
  const int n0   = blockIdx.x * 256 + wave * 16;
  const _Float16* Bg = Wx16_l + (size_t)b * 128 * 512;

  v16h bfr[4];
#pragma unroll
  for (int kt = 0; kt < 4; ++kt) bfr[kt] = load_b_frag_f16(Bg, 512, kt * 32, n0, lane);

  const int col  = n0 + (lane & 15);
  const int hl   = lane >> 4;
  const float bv = Bias_l[b * 512 + col];

  for (int mt = 0; mt < 8; ++mt) {             // M = T = 128 -> 8 tiles
    v8f c = {};
#pragma unroll
    for (int kt = 0; kt < 4; ++kt)
      c = wmma16(load_a_frag(Axs, 128, mt * 16, kt * 32, lane), bfr[kt], c);
#pragma unroll
    for (int v = 0; v < 8; ++v) {
      const int t = mt * 16 + hl * 8 + v;
      XG[((size_t)b * 128 + t) * 512 + col] = c[v] + bv;
    }
  }
}

// ---------------------------------------------------------------------------
// K4: recurrent scan.  One workgroup per example; Wh16[b] (128 KB, f16 pairs
// along k) pulled into LDS by the TDM, resident for all 128 steps.
// Thread j owns gate j.
__device__ __forceinline__ float sigf(float x) {
  return 1.0f / (1.0f + __expf(-x));
}

__global__ void k_scan(const float* __restrict__ XG,        // [64][128][512]
                       const _Float16* __restrict__ Wh16,   // [L][64][64][512][2]
                       const float* __restrict__ h0,        // [L][64][128]
                       const float* __restrict__ c0,        // [L][64][128]
                       float* __restrict__ Xout,            // [64][128][128]
                       float* __restrict__ hs,              // [L][64][128]
                       float* __restrict__ cs,              // [L][64][128]
                       int l) {
  __shared__ unsigned int Wp[64 * 512];        // 128 KB packed f16 pairs
  __shared__ float hbuf[128];
  __shared__ float gbuf[512];

  const int b   = blockIdx.x;
  const int tid = threadIdx.x;

  if ((tid >> 5) == 0) {                       // wave 0 issues the 128 KB DMA
    tdm_load_1d(Wp, Wh16 + (((size_t)l * 64 + b) * 64 * 512) * 2, 64 * 512 * 4);
    __builtin_amdgcn_s_wait_tensorcnt(0);
  }
  float cc = 0.0f;
  if (tid < 128) {
    hbuf[tid] = h0[((size_t)l * 64 + b) * 128 + tid];
    cc        = c0[((size_t)l * 64 + b) * 128 + tid];
  }
  __syncthreads();

  const float* xg = XG + (size_t)b * 128 * 512;

  for (int t = 0; t < 128; ++t) {
    // g[j] = XG[b][t][j] + sum_k h[k] * Wh[k][j]
    float acc = xg[t * 512 + tid];
#pragma unroll 8
    for (int kk = 0; kk < 64; ++kk) {
      union { unsigned int u; _Float16 h[2]; } cv;
      cv.u = Wp[kk * 512 + tid];                       // conflict-free
      const float2 hp = *(const float2*)&hbuf[2 * kk]; // broadcast
      acc = fmaf((float)cv.h[0], hp.x, acc);
      acc = fmaf((float)cv.h[1], hp.y, acc);
    }
    gbuf[tid] = acc;
    __syncthreads();
    if (tid < 128) {
      const float ig = gbuf[tid];
      const float fg = gbuf[128 + tid];
      const float ag = gbuf[256 + tid];
      const float og = gbuf[384 + tid];
      cc = cc * sigf(fg) + sigf(ig) * tanhf(ag);
      const float hn = tanhf(cc) * sigf(og);
      hbuf[tid] = hn;
      Xout[((size_t)b * 128 + t) * 128 + tid] = hn;
    }
    __syncthreads();
  }
  if (tid < 128) {
    hs[((size_t)l * 64 + b) * 128 + tid] = hbuf[tid];
    cs[((size_t)l * 64 + b) * 128 + tid] = cc;
  }
}

// ---------------------------------------------------------------------------
extern "C" void kernel_launch(void* const* d_in, const int* in_sizes, int n_in,
                              void* d_out, int out_size, void* d_ws, size_t ws_size,
                              hipStream_t stream) {
  (void)in_sizes; (void)n_in; (void)out_size; (void)ws_size;

  const float* input = (const float*)d_in[0];   // [64][128][128]
  const float* ph    = (const float*)d_in[1];   // [2][64][128]
  const float* pc    = (const float*)d_in[2];   // [2][64][128]
  const float* ctx   = (const float*)d_in[3];   // [64][512]
  const float* W1w   = (const float*)d_in[4];   // [2][512][512]
  const float* W2w   = (const float*)d_in[5];   // [2][512][131072]
  const float* W1b   = (const float*)d_in[6];   // [2][512][512]
  const float* W2b   = (const float*)d_in[7];   // [2][512][512]
  // d_in[8] = batch_sizes (constant == B, unused)

  char* ws = (char*)d_ws;
  _Float16* Ctx16 = (_Float16*)(ws + 0);                 //  65536 B
  _Float16* Hw16  = (_Float16*)(ws + 65536);             // 131072 B
  _Float16* Hb16  = (_Float16*)(ws + 196608);            // 131072 B
  float*    Bias  = (float*)   (ws + 327680);            // 262144 B
  _Float16* Wx16  = (_Float16*)(ws + 589824);            // 16 MB
  _Float16* Wh16  = (_Float16*)(ws + 17367040);          // 16 MB
  float*    XG    = (float*)   (ws + 34144256);          // 16 MB
  float*    Xbuf  = (float*)   (ws + 50921472);          //  4 MB

  float* x_out = (float*)d_out;                          // [64][128][128]
  float* hs    = x_out + 64 * 128 * 128;                 // [2][64][128]
  float* cs    = hs + 2 * 64 * 128;                      // [2][64][128]

  // K0: context -> f16
  k_cvt_f16<<<(BB * CTX + 255) / 256, 256, 0, stream>>>(ctx, Ctx16, BB * CTX);

  // K1: hypernet hidden layers + bias path
  for (int l = 0; l < 2; ++l) {
    k_small_gemm<true, true><<<8, 128, 0, stream>>>(
        Ctx16, W1w + (size_t)l * CTX * CH, (void*)(Hw16 + (size_t)l * BB * CH), CTX);
    k_small_gemm<true, true><<<8, 128, 0, stream>>>(
        Ctx16, W1b + (size_t)l * CTX * CH, (void*)(Hb16 + (size_t)l * BB * CH), CTX);
  }
  for (int l = 0; l < 2; ++l) {
    k_small_gemm<false, false><<<8, 128, 0, stream>>>(
        Hb16 + (size_t)l * BB * CH, W2b + (size_t)l * CH * GATES,
        (void*)(Bias + (size_t)l * BB * GATES), CH);
  }

  // K2: big weight generation (both layers)
  k_gen_w<<<dim3(1024, 2), 256, 0, stream>>>(Hw16, W2w, Wx16, Wh16);

  // Layer 0
  k_inproj<<<dim3(2, 64), 512, 0, stream>>>(
      input, Wx16, Bias, XG);
  k_scan<<<64, 512, 0, stream>>>(XG, Wh16, ph, pc, Xbuf, hs, cs, 0);

  // Layer 1
  k_inproj<<<dim3(2, 64), 512, 0, stream>>>(
      Xbuf, Wx16 + (size_t)1 * BB * 128 * 512, Bias + (size_t)1 * BB * GATES, XG);
  k_scan<<<64, 512, 0, stream>>>(XG, Wh16, ph, pc, x_out, hs, cs, 1);
}
